// QuantLeNet5_70514773066172
// MI455X (gfx1250) — 8-bit matrix operand order repaired and checked by reading, compiled, not run on hardware
//
#include <hip/hip_runtime.h>
#include <hip/hip_bf16.h>
#include <stdint.h>

typedef __attribute__((ext_vector_type(8))) int v8i;

__device__ __forceinline__ unsigned char qrelu2_u8(float v) {
  float y = fminf(fmaxf(v, 0.0f), 6.0f);
  return (unsigned char)(int)rintf(y * 0.5f);
}

__device__ __forceinline__ void bmap(int lane, int j, int* Klocal, int* N) {
  int v = j >> 2, b = j & 3;
  int laneHi = lane >> 4;
  *N = lane & 15;
  *Klocal = ((v >= 4) ? 32 : 0) + laneHi * 16 + (v & 3) * 4 + b;
}

__device__ __forceinline__ signed char sgn_i8(float w) {
  return (w > 0.f) ? (signed char)1 : ((w < 0.f) ? (signed char)-1 : (signed char)0);
}

__device__ __forceinline__ unsigned byteshift(unsigned hi, unsigned lo, int sh) {
#if __has_builtin(__builtin_amdgcn_alignbyte)
  return __builtin_amdgcn_alignbyte(hi, lo, (unsigned)sh);
#else
  return (sh == 0) ? lo : ((lo >> (8 * sh)) | (hi << (32 - 8 * sh)));
#endif
}

__device__ __forceinline__ v8i afrag_rowmajor(const unsigned char* src, int Mrow,
                                              int stride, int ktByte, int laneHi) {
  const unsigned char* base = src + Mrow * stride + ktByte + laneHi * 16;
  v8i a;
#pragma unroll
  for (int j = 0; j < 4; ++j) {
    unsigned long long q = *(const unsigned long long*)(base + (j >> 1) * 32 + (j & 1) * 8);
    a[2 * j]     = (int)(unsigned)q;
    a[2 * j + 1] = (int)(unsigned)(q >> 32);
  }
  return a;
}

__device__ __forceinline__ void async_g2l_b128(void* lds, const void* gmem) {
  unsigned la = (unsigned)(uintptr_t)lds;
  unsigned long long ga = (unsigned long long)(uintptr_t)gmem;
  asm volatile("global_load_async_to_lds_b128 %0, %1, off" :: "v"(la), "v"(ga) : "memory");
}
__device__ __forceinline__ void wait_async0() {
  asm volatile("s_wait_asynccnt 0" ::: "memory");
}

__global__ void k_prep(const float* __restrict__ w1, const float* __restrict__ w2,
                       const float* __restrict__ w3, const float* __restrict__ fw1,
                       const float* __restrict__ fw2, float* __restrict__ scales,
                       signed char* __restrict__ wB2, signed char* __restrict__ wB3,
                       signed char* __restrict__ wBf1, signed char* __restrict__ wBf2) {
  __shared__ float red[256];
  int tid = threadIdx.x;
  const float* ptrs[5] = {w1, w2, w3, fw1, fw2};
  const int cnts[5] = {150, 2400, 48000, 10080, 840};
  for (int t = 0; t < 5; ++t) {
    float sum = 0.f;
    for (int i = tid; i < cnts[t]; i += 256) sum += fabsf(ptrs[t][i]);
    red[tid] = sum;
    __syncthreads();
    for (int s = 128; s > 0; s >>= 1) {
      if (tid < s) red[tid] += red[tid + s];
      __syncthreads();
    }
    if (tid == 0) scales[t] = red[0] / (float)cnts[t];
    __syncthreads();
  }
  for (int idx = tid; idx < 4 * 1024; idx += 256) {
    int kt = idx >> 10, p = idx & 1023, lane = p >> 5, j = p & 31;
    int Kl, N; bmap(lane, j, &Kl, &N);
    int Kp = kt * 64 + Kl;
    int G = Kp >> 3, dx = Kp & 7;
    signed char v = 0;
    if (G < 30 && dx < 5) {
      int c = G / 5, dy = G % 5;
      v = sgn_i8(w2[N * 150 + c * 25 + dy * 5 + dx]);
    }
    wB2[idx] = v;
  }
  for (int idx = tid; idx < 8 * 7 * 1024; idx += 256) {
    int nt = idx / (7 * 1024);
    int rem = idx % (7 * 1024);
    int kt = rem >> 10, p = rem & 1023, lane = p >> 5, j = p & 31;
    int Kl, N; bmap(lane, j, &Kl, &N);
    int Ng = nt * 16 + N, K = kt * 64 + Kl;
    wB3[idx] = (Ng < 120 && K < 400) ? sgn_i8(w3[Ng * 400 + K]) : (signed char)0;
  }
  for (int idx = tid; idx < 6 * 2 * 1024; idx += 256) {
    int nt = idx / 2048;
    int rem = idx % 2048;
    int kt = rem >> 10, p = rem & 1023, lane = p >> 5, j = p & 31;
    int Kl, N; bmap(lane, j, &Kl, &N);
    int Ng = nt * 16 + N, K = kt * 64 + Kl;
    wBf1[idx] = (Ng < 84 && K < 120) ? sgn_i8(fw1[Ng * 120 + K]) : (signed char)0;
  }
  for (int idx = tid; idx < 2 * 1024; idx += 256) {
    int kt = idx >> 10, p = idx & 1023, lane = p >> 5, j = p & 31;
    int Kl, N; bmap(lane, j, &Kl, &N);
    int K = kt * 64 + Kl;
    wBf2[idx] = (N < 10 && K < 84) ? sgn_i8(fw2[N * 84 + K]) : (signed char)0;
  }
}

__global__ void k_conv1(const float* __restrict__ x, const float* __restrict__ w1,
                        const float* __restrict__ scales, unsigned char* __restrict__ h1p) {
  __shared__ __align__(16) float xs[1024];
  __shared__ float wq[152];
  int tid = threadIdx.x;
  int img = blockIdx.x;
  float s1 = scales[0];
  for (int i = tid; i < 256; i += 256)
    async_g2l_b128(&xs[i * 4], x + (size_t)img * 1024 + i * 4);
  if (tid < 150) {
    float w = w1[tid];
    wq[tid] = (w > 0.f) ? s1 : ((w < 0.f) ? -s1 : 0.f);
  }
  wait_async0();
  __syncthreads();
  for (int o = tid; o < 1176; o += 256) {
    int c = o / 196, p = o % 196, py = p / 14, px = p % 14;
    const float* wk = &wq[c * 25];
    float best = -1e30f;
    for (int dy = 0; dy < 2; ++dy)
      for (int dx = 0; dx < 2; ++dx) {
        int cy = 2 * py + dy, cx = 2 * px + dx;
        float acc = 0.f;
#pragma unroll
        for (int ky = 0; ky < 5; ++ky)
#pragma unroll
          for (int kx = 0; kx < 5; ++kx)
            acc = fmaf(xs[(cy + ky) * 32 + (cx + kx)], wk[ky * 5 + kx], acc);
        best = fmaxf(best, acc);
      }
    h1p[(size_t)img * 1680 + c * 280 + py * 20 + px] = qrelu2_u8(best);
  }
}

__global__ void k_conv2(const unsigned char* __restrict__ h1p,
                        const signed char* __restrict__ wB2,
                        const float* __restrict__ scales,
                        unsigned char* __restrict__ h2) {
  __shared__ __align__(16) unsigned char h1s[1680];
  __shared__ int cacc[112 * 16];
  int tid = threadIdx.x;
  int img = blockIdx.x;
  for (int i = tid; i < 105; i += 224)
    async_g2l_b128(&h1s[i * 16], h1p + (size_t)img * 1680 + i * 16);
  wait_async0();
  __syncthreads();

  int wave = tid >> 5, lane = tid & 31;
  int laneHi = lane >> 4;
  int rowA = wave * 16 + (lane & 15);
  int rp = (rowA >= 100) ? (rowA - 100) : rowA;
  int rowoff = (rp / 10) * 20 + (rp % 10);

  v8i bf[4], a[4];
#pragma unroll
  for (int kt = 0; kt < 4; ++kt)
    bf[kt] = *(const v8i*)(wB2 + kt * 1024 + lane * 32);
#pragma unroll
  for (int kt = 0; kt < 4; ++kt) {
#pragma unroll
    for (int j = 0; j < 4; ++j) {
      int G = kt * 8 + (j >> 1) * 4 + 2 * laneHi + (j & 1);
      int c = (G * 13) >> 6;
      int A = c * 280 + (G - c * 5) * 20 + rowoff;
      int Ab = A & ~3, sh = A & 3;
      unsigned d0 = *(const unsigned*)(h1s + Ab);
      unsigned d1 = *(const unsigned*)(h1s + Ab + 4);
      unsigned d2 = *(const unsigned*)(h1s + Ab + 8);
      a[kt][2 * j]     = (int)byteshift(d1, d0, sh);
      a[kt][2 * j + 1] = (int)byteshift(d2, d1, sh);
    }
  }
  v8i acc = {0, 0, 0, 0, 0, 0, 0, 0};
#pragma unroll
  for (int kt = 0; kt < 4; ++kt)
    acc = __builtin_amdgcn_wmma_i32_16x16x64_iu8(false, a[kt], true, bf[kt], acc, false, false);

#pragma unroll
  for (int r = 0; r < 8; ++r) {
    int M = r + 8 * laneHi;
    cacc[(wave * 16 + M) * 16 + (lane & 15)] = acc[r];
  }
  __syncthreads();
  float s2x2 = 2.f * scales[1];
  for (int o = tid; o < 400; o += 224) {
    int c = o / 25, pp = o % 25, py = pp / 5, px = pp % 5;
    int best = -2147483647;
#pragma unroll
    for (int d = 0; d < 2; ++d)
#pragma unroll
      for (int e = 0; e < 2; ++e) {
        int v = cacc[((2 * py + d) * 10 + (2 * px + e)) * 16 + c];
        best = (v > best) ? v : best;
      }
    h2[img * 400 + o] = qrelu2_u8((float)best * s2x2);
  }
}

__global__ void k_head(const unsigned char* __restrict__ h2,
                       const signed char* __restrict__ wB3,
                       const signed char* __restrict__ wBf1,
                       const signed char* __restrict__ wBf2,
                       const float* __restrict__ scales,
                       float* __restrict__ out, int B) {
  __shared__ __align__(16) unsigned char h2s[16 * 448];
  __shared__ __align__(16) unsigned char h3s[16 * 128];
  __shared__ __align__(16) unsigned char f1s[16 * 128];
  __shared__ float lg[16 * 16];
  int tid = threadIdx.x;
  int img0 = blockIdx.x * 16;
  for (int i = tid; i < 16 * 25; i += 256) {
    int r = i / 25, q = i % 25;
    if ((img0 + r) < B)
      async_g2l_b128(&h2s[r * 448 + q * 16], h2 + (size_t)(img0 + r) * 400 + q * 16);
  }
  wait_async0();
  __syncthreads();
  int wave = tid >> 5, lane = tid & 31;
  int laneHi = lane >> 4, Mrow = lane & 15, Ncol = lane & 15;

  {
    float s = 2.f * scales[2];
    v8i acc = {0, 0, 0, 0, 0, 0, 0, 0};
    {
      v8i a[4], bf[4];
#pragma unroll
      for (int kt = 0; kt < 4; ++kt) {
        bf[kt] = *(const v8i*)(wB3 + (wave * 7 + kt) * 1024 + lane * 32);
        a[kt] = afrag_rowmajor(h2s, Mrow, 448, kt * 64, laneHi);
      }
#pragma unroll
      for (int kt = 0; kt < 4; ++kt)
        acc = __builtin_amdgcn_wmma_i32_16x16x64_iu8(false, a[kt], true, bf[kt], acc, false, false);
    }
    {
      v8i a[3], bf[3];
#pragma unroll
      for (int kt = 0; kt < 3; ++kt) {
        bf[kt] = *(const v8i*)(wB3 + (wave * 7 + 4 + kt) * 1024 + lane * 32);
        a[kt] = afrag_rowmajor(h2s, Mrow, 448, (4 + kt) * 64, laneHi);
      }
#pragma unroll
      for (int kt = 0; kt < 3; ++kt)
        acc = __builtin_amdgcn_wmma_i32_16x16x64_iu8(false, a[kt], true, bf[kt], acc, false, false);
    }
#pragma unroll
    for (int r = 0; r < 8; ++r) {
      int M = r + 8 * laneHi, ch = wave * 16 + Ncol;
      if (ch < 120) h3s[M * 128 + ch] = qrelu2_u8((float)acc[r] * s);
    }
  }
  __syncthreads();

  if (wave < 6) {
    float s = 2.f * scales[3];
    v8i acc = {0, 0, 0, 0, 0, 0, 0, 0};
    v8i a[2], bf[2];
#pragma unroll
    for (int kt = 0; kt < 2; ++kt) {
      bf[kt] = *(const v8i*)(wBf1 + (wave * 2 + kt) * 1024 + lane * 32);
      a[kt] = afrag_rowmajor(h3s, Mrow, 128, kt * 64, laneHi);
    }
#pragma unroll
    for (int kt = 0; kt < 2; ++kt)
      acc = __builtin_amdgcn_wmma_i32_16x16x64_iu8(false, a[kt], true, bf[kt], acc, false, false);
#pragma unroll
    for (int r = 0; r < 8; ++r) {
      int M = r + 8 * laneHi, ch = wave * 16 + Ncol;
      if (ch < 96) f1s[M * 128 + ch] = qrelu2_u8((float)acc[r] * s);
    }
  }
  __syncthreads();

  if (wave == 0) {
    float s = 2.f * scales[4];
    v8i acc = {0, 0, 0, 0, 0, 0, 0, 0};
    v8i a[2], bf[2];
#pragma unroll
    for (int kt = 0; kt < 2; ++kt) {
      bf[kt] = *(const v8i*)(wBf2 + kt * 1024 + lane * 32);
      a[kt] = afrag_rowmajor(f1s, Mrow, 128, kt * 64, laneHi);
    }
#pragma unroll
    for (int kt = 0; kt < 2; ++kt)
      acc = __builtin_amdgcn_wmma_i32_16x16x64_iu8(false, a[kt], true, bf[kt], acc, false, false);
#pragma unroll
    for (int r = 0; r < 8; ++r) lg[(r + 8 * laneHi) * 16 + Ncol] = (float)acc[r] * s;
  }
  __syncthreads();

  if (tid < 16 && (img0 + tid) < B) {
    float m = lg[tid * 16];
    for (int j = 1; j < 10; ++j) m = fmaxf(m, lg[tid * 16 + j]);
    float e[10], sum = 0.f;
    for (int j = 0; j < 10; ++j) { e[j] = expf(lg[tid * 16 + j] - m); sum += e[j]; }
    float inv = 1.f / sum;
    for (int j = 0; j < 10; ++j) out[(size_t)(img0 + tid) * 10 + j] = e[j] * inv;
  }
}

extern "C" void kernel_launch(void* const* d_in, const int* in_sizes, int n_in,
                              void* d_out, int out_size, void* d_ws, size_t ws_size,
                              hipStream_t stream) {
  const float* x   = (const float*)d_in[0];
  const float* w1  = (const float*)d_in[1];
  const float* w2  = (const float*)d_in[2];
  const float* w3  = (const float*)d_in[3];
  const float* fw1 = (const float*)d_in[4];
  const float* fw2 = (const float*)d_in[5];
  float* out = (float*)d_out;
  int B = in_sizes[0] / 1024;

  char* ws = (char*)d_ws;
  float*       scales = (float*)ws;
  signed char* wB2  = (signed char*)(ws + 256);
  signed char* wB3  = (signed char*)(ws + 4352);
  signed char* wBf1 = (signed char*)(ws + 61696);
  signed char* wBf2 = (signed char*)(ws + 73984);
  unsigned char* h1p = (unsigned char*)(ws + 76032);
  unsigned char* h2  = h1p + (size_t)B * 1680;

  k_prep<<<1, 256, 0, stream>>>(w1, w2, w3, fw1, fw2, scales, wB2, wB3, wBf1, wBf2);
  k_conv1<<<B, 256, 0, stream>>>(x, w1, scales, h1p);
  k_conv2<<<B, 224, 0, stream>>>(h1p, wB2, scales, h2);
  k_head<<<(B + 15) / 16, 256, 0, stream>>>(h2, wB3, wBf1, wBf2, scales, out, B);
}
